// FuzzyMultiheadAttention_64080912056504
// MI455X (gfx1250) — matrix operand order repaired and checked by reading, not compiled, not run on hardware
//
#include <hip/hip_runtime.h>
#include <hip/hip_bf16.h>

#define EMBED 512
#define NHEAD 8
#define NRULE 16
#define HDIM  64

typedef __attribute__((ext_vector_type(16))) _Float16 v16h;
typedef __attribute__((ext_vector_type(8)))  _Float16 h8;
typedef __attribute__((ext_vector_type(8)))  float    v8f;

// ---------------------------------------------------------------------------
// WMMA fragment loaders for V_WMMA_F32_16X16X32_F16 (wave32; contiguous k order, used for both A and B)
// A: 16(M)x32(K). lanes 0-15: M=lane, K = 0..15; lanes 16-31: K = 16..31.
// B: 32(K)x16(N). lane&15 = N (a row of W, since B = W^T);
//    lanes 0-15 hold K=0..15, lanes 16-31 hold K=16..31 (contiguous in W row).
// ---------------------------------------------------------------------------
static __device__ __forceinline__ v16h load_a_frag(const _Float16* __restrict__ base,
                                                   int ld, int lane) {
  const int m  = lane & 15;
  const int kb = (lane & 16) ? 16 : 0;
  const _Float16* p = base + (size_t)m * ld + kb;
  const h8 lo = *(const h8*)(p);
  const h8 hi = *(const h8*)(p + 8);
  v16h a;
#pragma unroll
  for (int j = 0; j < 8; ++j) { a[j] = lo[j]; a[8 + j] = hi[j]; }
  return a;
}

static __device__ __forceinline__ v16h load_b_frag(const _Float16* __restrict__ wtile,
                                                   int ld, int lane) {
  const int n  = lane & 15;
  const int kb = (lane & 16) ? 16 : 0;
  const _Float16* p = wtile + (size_t)n * ld + kb;
  const h8 lo = *(const h8*)(p);
  const h8 hi = *(const h8*)(p + 8);
  v16h b;
#pragma unroll
  for (int j = 0; j < 8; ++j) { b[j] = lo[j]; b[8 + j] = hi[j]; }
  return b;
}

// ---------------------------------------------------------------------------
// One-time f32 -> f16 staging (weights reused up to 512x; keep cvt off hot loop)
// ---------------------------------------------------------------------------
__global__ void cvt_f32_f16_kernel(const float* __restrict__ s,
                                   _Float16* __restrict__ d, size_t n) {
  size_t i = (size_t)blockIdx.x * blockDim.x + threadIdx.x;
  const size_t stride = (size_t)gridDim.x * blockDim.x;
  for (; i < n; i += stride) d[i] = (_Float16)s[i];
}

// ---------------------------------------------------------------------------
// C = (A @ W^T + bias) * alpha.   A: MxK f16, W: NxK f16 (row-major), C: MxN.
// Wave tile: 32(M) x 64(N): per K-step 2 A-frags + 4 B-frags feed 8 WMMAs
// (0.75 fragment loads per WMMA; B reused across both row blocks).
// ---------------------------------------------------------------------------
template <typename OutT>
__global__ __launch_bounds__(256) void gemm_bias_kernel(
    const _Float16* __restrict__ A, const _Float16* __restrict__ W,
    const float* __restrict__ bias, OutT* __restrict__ C,
    int M, int N, int K, float alpha) {
  const int lane = threadIdx.x & 31;
  const int wave = blockIdx.x * (blockDim.x >> 5) + (threadIdx.x >> 5);
  const int ncol = N >> 6;
  const int mi = wave / ncol;
  const int nj = wave % ncol;
  if (mi * 32 >= M) return;                    // wave-uniform guard
  const _Float16* Arow0 = A + (size_t)(mi * 32) * K;
  const _Float16* Arow1 = Arow0 + (size_t)16 * K;
  const _Float16* W0    = W + (size_t)(nj * 64) * K;
  v8f c0[4] = {};
  v8f c1[4] = {};
  for (int k0 = 0; k0 < K; k0 += 32) {
    const v16h a0 = load_a_frag(Arow0 + k0, K, lane);
    const v16h a1 = load_a_frag(Arow1 + k0, K, lane);
#pragma unroll
    for (int t = 0; t < 4; ++t) {
      const v16h b = load_b_frag(W0 + (size_t)t * 16 * K + k0, K, lane);
      c0[t] = __builtin_amdgcn_wmma_f32_16x16x32_f16(false, a0, false, b,
                                                     (short)0, c0[t], false, false);
      c1[t] = __builtin_amdgcn_wmma_f32_16x16x32_f16(false, a1, false, b,
                                                     (short)0, c1[t], false, false);
    }
  }
  const int n = lane & 15;
  const int rowoff = (lane & 16) ? 8 : 0;
#pragma unroll
  for (int t = 0; t < 4; ++t) {
    const int col = nj * 64 + t * 16 + n;
    const float bb = bias[col];
#pragma unroll
    for (int i = 0; i < 8; ++i) {
      const int row0 = mi * 32 + rowoff + i;
      C[(size_t)row0 * N + col] = (OutT)(alpha * (c0[t][i] + bb));
      C[(size_t)(row0 + 16) * N + col] = (OutT)(alpha * (c1[t][i] + bb));
    }
  }
}

// ---------------------------------------------------------------------------
// Fuzzy rule scores + softmax over R=16. 16 lanes = 16 rules per (token,head);
// cross-lane max/sum via half-wave shuffles (wave32).
// ---------------------------------------------------------------------------
__global__ __launch_bounds__(256) void rules_attn_kernel(
    const _Float16* __restrict__ q, const float* __restrict__ rk,
    const float* __restrict__ rw, float* __restrict__ attn, int tokens) {
  const int tid = blockIdx.x * 256 + threadIdx.x;
  const int g = tid >> 4;                       // (token, head) group
  const int r = tid & 15;
  if (g >= tokens * NHEAD) return;              // uniform per 16-lane group
  const int token = g >> 3;
  const int h = g & 7;
  const _Float16* qrow = q + (size_t)token * EMBED + h * HDIM;
  const float* krow = rk + ((size_t)h * NRULE + r) * HDIM;
  const float* wrow = rw + ((size_t)h * NRULE + r) * HDIM;
  float acc = 0.f;
#pragma unroll 8
  for (int d = 0; d < HDIM; ++d) {
    const float df = ((float)qrow[d] - krow[d]) / wrow[d];
    acc += df * df;
  }
  const float z = -0.5f * (acc * (1.0f / HDIM));
  float m = z;
  m = fmaxf(m, __shfl_xor(m, 8, 32));
  m = fmaxf(m, __shfl_xor(m, 4, 32));
  m = fmaxf(m, __shfl_xor(m, 2, 32));
  m = fmaxf(m, __shfl_xor(m, 1, 32));
  const float e = __expf(z - m);
  float s = e;
  s += __shfl_xor(s, 8, 32);
  s += __shfl_xor(s, 4, 32);
  s += __shfl_xor(s, 2, 32);
  s += __shfl_xor(s, 1, 32);
  attn[(size_t)g * NRULE + r] = e / s;
}

// ---------------------------------------------------------------------------
// Fused v-projection + rule contraction (never materializes the 268 MB v).
// GEMM: tokens x (E*R) x E.  A 16-column WMMA tile == one (h,d) x all 16 rules,
// so sum_r attn*(C+bv)*scale is a 16-lane shuffle reduction in the epilogue.
// Wave tile 32(M) x 64(N), same 8-WMMA K-step as gemm_bias_kernel.
// ---------------------------------------------------------------------------
__global__ __launch_bounds__(256) void vproj_rules_kernel(
    const _Float16* __restrict__ Vh, const _Float16* __restrict__ Wv,
    const float* __restrict__ bv, const float* __restrict__ attn,
    _Float16* __restrict__ outpre, int M, int K, float scale) {
  const int lane = threadIdx.x & 31;
  const int wave = blockIdx.x * (blockDim.x >> 5) + (threadIdx.x >> 5);
  const int ncol = (EMBED * NRULE) >> 6;        // 128 wave-column tiles
  const int mi = wave / ncol;
  const int nj = wave % ncol;
  if (mi * 32 >= M) return;
  const _Float16* Arow0 = Vh + (size_t)(mi * 32) * K;
  const _Float16* Arow1 = Arow0 + (size_t)16 * K;
  const _Float16* W0    = Wv + (size_t)(nj * 64) * K;
  v8f c0[4] = {};
  v8f c1[4] = {};
  for (int k0 = 0; k0 < K; k0 += 32) {
    const v16h a0 = load_a_frag(Arow0 + k0, K, lane);
    const v16h a1 = load_a_frag(Arow1 + k0, K, lane);
#pragma unroll
    for (int t = 0; t < 4; ++t) {
      const v16h b = load_b_frag(W0 + (size_t)t * 16 * K + k0, K, lane);
      c0[t] = __builtin_amdgcn_wmma_f32_16x16x32_f16(false, a0, false, b,
                                                     (short)0, c0[t], false, false);
      c1[t] = __builtin_amdgcn_wmma_f32_16x16x32_f16(false, a1, false, b,
                                                     (short)0, c1[t], false, false);
    }
  }
  const int r = lane & 15;                      // N within tile == rule index
  const int rowoff = (lane & 16) ? 8 : 0;
#pragma unroll
  for (int t = 0; t < 4; ++t) {
    const int colbase = nj * 64 + t * 16;       // = (h*64+d)*16, r-aligned
    const int hd = colbase >> 4;                // h*64 + d
    const int h  = hd >> 6;
    const float bb = bv[colbase + r];
    float red0[8], red1[8];
#pragma unroll
    for (int i = 0; i < 8; ++i) {
      const int token0 = mi * 32 + rowoff + i;
      const int token1 = token0 + 16;
      const float aw0 = attn[((size_t)token0 * NHEAD + h) * NRULE + r];
      const float aw1 = attn[((size_t)token1 * NHEAD + h) * NRULE + r];
      float v0 = scale * (c0[t][i] + bb) * aw0;
      float v1 = scale * (c1[t][i] + bb) * aw1;
      v0 += __shfl_xor(v0, 8, 32);              // reduce over 16 rule lanes
      v0 += __shfl_xor(v0, 4, 32);
      v0 += __shfl_xor(v0, 2, 32);
      v0 += __shfl_xor(v0, 1, 32);
      v1 += __shfl_xor(v1, 8, 32);
      v1 += __shfl_xor(v1, 4, 32);
      v1 += __shfl_xor(v1, 2, 32);
      v1 += __shfl_xor(v1, 1, 32);
      red0[i] = v0;
      red1[i] = v1;
    }
    if (r == 0) {
#pragma unroll
      for (int i = 0; i < 8; ++i) {
        const int token0 = mi * 32 + rowoff + i;
        outpre[(size_t)token0 * EMBED + hd] = (_Float16)red0[i];
        outpre[(size_t)(token0 + 16) * EMBED + hd] = (_Float16)red1[i];
      }
    }
  }
}

// ---------------------------------------------------------------------------
extern "C" void kernel_launch(void* const* d_in, const int* in_sizes, int n_in,
                              void* d_out, int out_size, void* d_ws, size_t ws_size,
                              hipStream_t stream) {
  const float* query = (const float*)d_in[0];
  // d_in[1] = key : unused by the reference
  const float* value = (const float*)d_in[2];
  const float* Wq = (const float*)d_in[3];
  const float* bq = (const float*)d_in[4];
  const float* Wv = (const float*)d_in[5];
  const float* bv = (const float*)d_in[6];
  const float* Wo = (const float*)d_in[7];
  const float* bo = (const float*)d_in[8];
  const float* rk = (const float*)d_in[9];
  const float* rw = (const float*)d_in[10];
  float* out = (float*)d_out;

  const int tokens = in_sizes[0] / EMBED;       // B*S = 8192
  const size_t tE = (size_t)tokens * EMBED;
  const float scale = 0.125f;                   // D^-0.5 = 1/8

  // workspace layout (f16 staging + f32 attn), ~37 MB total
  _Float16* query_h  = (_Float16*)d_ws;
  _Float16* value_h  = query_h + tE;
  _Float16* Wq_h     = value_h + tE;
  _Float16* Wv_h     = Wq_h + (size_t)EMBED * EMBED;
  _Float16* Wo_h     = Wv_h + (size_t)EMBED * EMBED * NRULE;
  _Float16* q_h      = Wo_h + (size_t)EMBED * EMBED;
  float*    attn     = (float*)(q_h + tE);
  _Float16* outpre_h = query_h;                 // alias: query_h dead after k1

  // one-time f32 -> f16 staging
  cvt_f32_f16_kernel<<<2048, 256, 0, stream>>>(query, query_h, tE);
  cvt_f32_f16_kernel<<<2048, 256, 0, stream>>>(value, value_h, tE);
  cvt_f32_f16_kernel<<<512,  256, 0, stream>>>(Wq, Wq_h, (size_t)EMBED * EMBED);
  cvt_f32_f16_kernel<<<4096, 256, 0, stream>>>(Wv, Wv_h, (size_t)EMBED * EMBED * NRULE);
  cvt_f32_f16_kernel<<<512,  256, 0, stream>>>(Wo, Wo_h, (size_t)EMBED * EMBED);

  // 1) q = (query @ Wq^T + bq) * scale            [WMMA GEMM]
  {
    const int waves = (tokens / 32) * (EMBED / 64);   // 2048
    gemm_bias_kernel<_Float16><<<waves / 8, 256, 0, stream>>>(
        query_h, Wq_h, bq, q_h, tokens, EMBED, EMBED, scale);
  }
  // 2) fuzzy z-scores + softmax over rules
  {
    const int threads = tokens * NHEAD * NRULE;       // 1,048,576
    rules_attn_kernel<<<threads / 256, 256, 0, stream>>>(q_h, rk, rw, attn, tokens);
  }
  // 3) out_pre = sum_r attn * ((value @ Wv^T + bv) * scale)   [fused WMMA GEMM]
  {
    const int waves = (tokens / 32) * ((EMBED * NRULE) / 64);  // 32768
    vproj_rules_kernel<<<waves / 8, 256, 0, stream>>>(
        value_h, Wv_h, bv, attn, outpre_h, tokens, EMBED, scale);
  }
  // 4) out = out_pre @ Wo^T + bo                   [WMMA GEMM]
  {
    const int waves = (tokens / 32) * (EMBED / 64);
    gemm_bias_kernel<float><<<waves / 8, 256, 0, stream>>>(
        outpre_h, Wo_h, bo, out, tokens, EMBED, EMBED, 1.0f);
  }
}
